// AutoRegressive_7241314861734
// MI455X (gfx1250) — matrix operand order repaired and checked by reading, not compiled, not run on hardware
//
#include <hip/hip_runtime.h>

// Problem constants (from reference)
#define B_BATCH   2048
#define T_WARM    128
#define FEAT      64
#define UNITS     512
#define OUT_STEPS 32
#define GATES4    (4 * UNITS)   // 2048

// LDS B-tile: 256 gate-columns x 32 K, padded column stride to kill bank conflicts
#define BCOLS      256
#define COL_STRIDE 40           // halves; 80B -> stride-20-dword walk, conflict-free b128

typedef __attribute__((ext_vector_type(16))) _Float16 v16h;
typedef __attribute__((ext_vector_type(8)))  _Float16 v8h;
typedef __attribute__((ext_vector_type(8)))  float    v8f;
typedef __attribute__((ext_vector_type(4)))  int      v4i;

union AFrag {
    v16h v;
    v8h  h[2];
};

__device__ __forceinline__ float sigmoidf_(float x) {
    return 1.0f / (1.0f + __expf(-x));
}

// Async global->LDS path (gfx1250 ASYNCcnt) when the toolchain exposes it.
#if defined(__has_builtin)
#if __has_builtin(__builtin_amdgcn_global_load_async_to_lds_b128) && \
    __has_builtin(__builtin_amdgcn_s_wait_asynccnt)
#define USE_ASYNC_LDS 1
#endif
#endif
#ifndef USE_ASYNC_LDS
#define USE_ASYNC_LDS 0
#endif

#if USE_ASYNC_LDS
typedef __attribute__((address_space(1))) v4i as1_v4i;   // global
typedef __attribute__((address_space(3))) v4i as3_v4i;   // LDS
#endif

// ---------------------------------------------------------------------------
// One-time prep: transpose weights to K-major f16, zero h and c state.
// ---------------------------------------------------------------------------
__global__ void __launch_bounds__(256)
prep_kernel(const float* __restrict__ W_x, const float* __restrict__ W_h,
            const float* __restrict__ dense_W,
            _Float16* __restrict__ Wt_x, _Float16* __restrict__ Wt_h,
            _Float16* __restrict__ Wt_d,
            _Float16* __restrict__ h0, float* __restrict__ c0)
{
    int idx = blockIdx.x * blockDim.x + threadIdx.x;   // [0, 2048*512)
    if (idx < GATES4 * UNITS) {
        int n = idx / UNITS;
        int k = idx % UNITS;
        Wt_h[idx] = (_Float16)W_h[(size_t)k * GATES4 + n];
    }
    if (idx < B_BATCH * UNITS) {
        h0[idx] = (_Float16)0.0f;
        c0[idx] = 0.0f;
    }
    if (idx < GATES4 * FEAT) {
        int n = idx / FEAT;
        int k = idx % FEAT;
        Wt_x[idx] = (_Float16)W_x[(size_t)k * GATES4 + n];
    }
    if (idx < FEAT * UNITS) {
        int n = idx / UNITS;
        int k = idx % UNITS;
        Wt_d[idx] = (_Float16)dense_W[(size_t)k * FEAT + n];
    }
}

// ---------------------------------------------------------------------------
// Fused LSTM step with LDS-staged, double-buffered B panel.
// Grid: (B/128, U/64), 256 threads (8 waves). Wave owns 16 batch rows and
// all 4 gates x 64 units = 16 f32 accumulator tiles of 16x16.
// XSRC: 0 = x from f32 inputs[:, t, :], 1 = x from f16 pred buffer [B,64].
// ---------------------------------------------------------------------------
template <int XSRC>
__global__ void __launch_bounds__(256)
lstm_step_kernel(const _Float16* __restrict__ h_in,
                 _Float16* __restrict__ h_out,
                 float* __restrict__ c,
                 const void* __restrict__ xsrc, int t,
                 const _Float16* __restrict__ Wt_h,
                 const _Float16* __restrict__ Wt_x,
                 const float* __restrict__ bias)
{
    __shared__ _Float16 Bs[2][BCOLS * COL_STRIDE];   // 2 x 20 KB

    const int tid   = threadIdx.x;
    const int lane  = tid & 31;
    const int wave  = tid >> 5;
    const int rBase = blockIdx.x * 128 + wave * 16;   // batch rows for this wave
    const int uBase = blockIdx.y * 64;                // unit block
    const int lmod  = lane & 15;
    const int lhi   = lane >> 4;                      // 0 or 1
    const int arow  = rBase + lmod;                   // A-matrix row for this lane

    // Cooperative stage of one 256x32 B tile (thread tid owns column tid).
    auto stage_tile = [&](int bsel, const _Float16* __restrict__ src, int kstride, int k0) {
        const int gate = tid >> 6;
        const int ncol = gate * UNITS + uBase + (tid & 63);
        const _Float16* g = src + (size_t)ncol * kstride + k0;
        _Float16* l = &Bs[bsel][tid * COL_STRIDE];
#if USE_ASYNC_LDS
#pragma unroll
        for (int j = 0; j < 4; ++j)
            __builtin_amdgcn_global_load_async_to_lds_b128(
                (as1_v4i*)(g + j * 8),
                (as3_v4i*)(l + j * 8), 0, 0);
#else
        v8h tmp[4];
#pragma unroll
        for (int j = 0; j < 4; ++j) tmp[j] = *(const v8h*)(g + j * 8);
#pragma unroll
        for (int j = 0; j < 4; ++j) *(v8h*)(l + j * 8) = tmp[j];
#endif
    };
    auto stage_wait = [&]() {
#if USE_ASYNC_LDS
        __builtin_amdgcn_s_wait_asynccnt(0);
#endif
        __syncthreads();
    };

    v8f zero = {};
    v8f acc[16];
#pragma unroll
    for (int i = 0; i < 16; ++i) acc[i] = zero;

    // 16 WMMAs against the LDS-resident B tile.
    auto mma16 = [&](int bsel, const AFrag& a) {
#pragma unroll
        for (int nt = 0; nt < 16; ++nt) {
            const int cc = (nt >> 2) * 64 + (nt & 3) * 16 + lmod;
            const _Float16* bp = &Bs[bsel][cc * COL_STRIDE + lhi * 8];
            AFrag bf;
            bf.h[0] = *(const v8h*)(bp);
            bf.h[1] = *(const v8h*)(bp + 16);
            acc[nt] = __builtin_amdgcn_wmma_f32_16x16x32_f16(
                false, a.v, false, bf.v, (short)0, acc[nt], false, false);
        }
    };

    // ---- Pipeline prologue: stage first h-weight tile ----
    stage_tile(0, Wt_h, UNITS, 0);
    stage_wait();
    int buf = 0;

    // ---- K loop over hidden state (K = 512) ----
    for (int i = 0; i < UNITS / 32; ++i) {
        if (i < UNITS / 32 - 1) stage_tile(buf ^ 1, Wt_h, UNITS, (i + 1) * 32);
        else                    stage_tile(buf ^ 1, Wt_x, FEAT, 0);   // first x tile

        AFrag a;
        const _Float16* ap = h_in + (size_t)arow * UNITS + i * 32 + lhi * 8;
        a.h[0] = *(const v8h*)(ap);        // K = k0 + lhi*8 .. +7
        a.h[1] = *(const v8h*)(ap + 16);   // K = k0 + 16 + lhi*8 .. +7

        mma16(buf, a);
        stage_wait();
        buf ^= 1;
    }

    // ---- K loop over input features (K = 64, 2 tiles) ----
    for (int i = 0; i < 2; ++i) {
        if (i == 0) stage_tile(buf ^ 1, Wt_x, FEAT, 32);

        AFrag a;
        if (XSRC == 0) {
            const float* xp = (const float*)xsrc +
                              ((size_t)arow * T_WARM + t) * FEAT + i * 32;
#pragma unroll
            for (int g = 0; g < 2; ++g) {
                const float* p = xp + g * 16 + lhi * 8;
#pragma unroll
                for (int j = 0; j < 8; ++j)
                    a.h[g][j] = (_Float16)p[j];
            }
        } else {
            const _Float16* xp = (const _Float16*)xsrc +
                                 (size_t)arow * FEAT + i * 32 + lhi * 8;
            a.h[0] = *(const v8h*)(xp);
            a.h[1] = *(const v8h*)(xp + 16);
        }

        mma16(buf, a);
        if (i == 0) { stage_wait(); buf ^= 1; }
    }

    // ---- Gate combine: same (row,col) lives at same (VGPR,lane) in each tile.
#pragma unroll
    for (int un = 0; un < 4; ++un) {
        const int ucol = uBase + un * 16 + lmod;
        const float bi  = bias[ucol];
        const float bf_ = bias[UNITS + ucol];
        const float bg  = bias[2 * UNITS + ucol];
        const float bo  = bias[3 * UNITS + ucol];
#pragma unroll
        for (int r = 0; r < 8; ++r) {
            const int row = rBase + r + lhi * 8;
            const float zi = acc[un][r] + bi;
            const float zf = acc[4 + un][r] + bf_;
            const float zg = acc[8 + un][r] + bg;
            const float zo = acc[12 + un][r] + bo;
            const float ig = sigmoidf_(zi);
            const float fg = sigmoidf_(zf);
            const float gg = tanhf(zg);
            const float og = sigmoidf_(zo);
            const size_t cidx = (size_t)row * UNITS + ucol;
            const float cn = fg * c[cidx] + ig * gg;
            c[cidx] = cn;
            const float hn = og * tanhf(cn);
            h_out[cidx] = (_Float16)hn;
        }
    }
}

// ---------------------------------------------------------------------------
// Dense projection: pred = h @ dense_W + dense_b  ([2048x512]@[512x64]).
// Small (1/36 of a step's FLOPs); direct-global loads are fine here.
// ---------------------------------------------------------------------------
__global__ void __launch_bounds__(256)
dense_kernel(const _Float16* __restrict__ h16,
             const _Float16* __restrict__ Wt_d,
             const float* __restrict__ dense_b,
             float* __restrict__ out, int step,
             _Float16* __restrict__ pred16)
{
    const int lane  = threadIdx.x & 31;
    const int wave  = threadIdx.x >> 5;
    const int rBase = blockIdx.x * 128 + wave * 16;
    const int lmod  = lane & 15;
    const int lhi   = lane >> 4;

    v8f zero = {};
    v8f acc[4];
#pragma unroll
    for (int i = 0; i < 4; ++i) acc[i] = zero;

    const int arow = rBase + lmod;

    for (int k0 = 0; k0 < UNITS; k0 += 32) {
        AFrag a;
        const _Float16* ap = h16 + (size_t)arow * UNITS + k0 + lhi * 8;
        a.h[0] = *(const v8h*)(ap);
        a.h[1] = *(const v8h*)(ap + 16);
#pragma unroll
        for (int nt = 0; nt < 4; ++nt) {
            const int ncol = nt * 16 + lmod;
            const _Float16* bp = Wt_d + (size_t)ncol * UNITS + k0 + lhi * 8;
            AFrag bf;
            bf.h[0] = *(const v8h*)(bp);
            bf.h[1] = *(const v8h*)(bp + 16);
            acc[nt] = __builtin_amdgcn_wmma_f32_16x16x32_f16(
                false, a.v, false, bf.v, (short)0, acc[nt], false, false);
        }
    }

#pragma unroll
    for (int nt = 0; nt < 4; ++nt) {
        const int col = nt * 16 + lmod;
        const float bb = dense_b[col];
#pragma unroll
        for (int r = 0; r < 8; ++r) {
            const int row = rBase + r + lhi * 8;
            const float v = acc[nt][r] + bb;
            out[(size_t)row * OUT_STEPS * FEAT + (size_t)step * FEAT + col] = v;
            pred16[(size_t)row * FEAT + col] = (_Float16)v;
        }
    }
}

// ---------------------------------------------------------------------------
extern "C" void kernel_launch(void* const* d_in, const int* in_sizes, int n_in,
                              void* d_out, int out_size, void* d_ws, size_t ws_size,
                              hipStream_t stream)
{
    (void)in_sizes; (void)n_in; (void)out_size; (void)ws_size;
    const float* inputs  = (const float*)d_in[0];
    const float* W_x     = (const float*)d_in[1];
    const float* W_h     = (const float*)d_in[2];
    const float* bias    = (const float*)d_in[3];
    const float* dense_W = (const float*)d_in[4];
    const float* dense_b = (const float*)d_in[5];
    float* out = (float*)d_out;

    char* ws = (char*)d_ws;
    size_t off = 0;
    auto take = [&](size_t bytes) -> void* {
        void* p = ws + off;
        off += (bytes + 255) & ~(size_t)255;
        return p;
    };

    _Float16* Wt_h   = (_Float16*)take((size_t)GATES4 * UNITS * 2);  // 2 MB
    _Float16* Wt_x   = (_Float16*)take((size_t)GATES4 * FEAT * 2);   // 256 KB
    _Float16* Wt_d   = (_Float16*)take((size_t)FEAT * UNITS * 2);    // 64 KB
    _Float16* hA     = (_Float16*)take((size_t)B_BATCH * UNITS * 2); // 2 MB
    _Float16* hB     = (_Float16*)take((size_t)B_BATCH * UNITS * 2); // 2 MB
    float*    cbuf   = (float*)take((size_t)B_BATCH * UNITS * 4);    // 4 MB
    _Float16* pred16 = (_Float16*)take((size_t)B_BATCH * FEAT * 2);  // 256 KB

    prep_kernel<<<(GATES4 * UNITS) / 256, 256, 0, stream>>>(
        W_x, W_h, dense_W, Wt_x, Wt_h, Wt_d, hA, cbuf);

    dim3 grid(B_BATCH / 128, UNITS / 64);
    _Float16* hin  = hA;
    _Float16* hout = hB;

    // Warm-up: 128 sequential LSTM steps over the input sequence.
    for (int t = 0; t < T_WARM; ++t) {
        lstm_step_kernel<0><<<grid, 256, 0, stream>>>(
            hin, hout, cbuf, (const void*)inputs, t, Wt_h, Wt_x, bias);
        _Float16* tmp = hin; hin = hout; hout = tmp;
    }

    // First prediction.
    dense_kernel<<<B_BATCH / 128, 256, 0, stream>>>(
        hin, Wt_d, dense_b, out, 0, pred16);

    // Autoregressive decode: 31 more steps feeding pred back as x.
    for (int s = 1; s < OUT_STEPS; ++s) {
        lstm_step_kernel<1><<<grid, 256, 0, stream>>>(
            hin, hout, cbuf, (const void*)pred16, 0, Wt_h, Wt_x, bias);
        _Float16* tmp = hin; hin = hout; hout = tmp;
        dense_kernel<<<B_BATCH / 128, 256, 0, stream>>>(
            hin, Wt_d, dense_b, out, s, pred16);
    }
}
